// LlamaAttention_heavy_hitter_55353538511456
// MI455X (gfx1250) — matrix operand order repaired and checked by reading, not compiled, not run on hardware
//
#include <hip/hip_runtime.h>
#include <hip/hip_bf16.h>

// ---------------- problem constants ----------------
#define S_LEN   2048
#define DMODEL  2048
#define NH      16
#define HDIM    128
#define CACHE   408      // heavy(204)+recent(204)
#define RECENT  204
#define PENALTY_F 0.4f
#define NEG_MIN_F (-3.4028234663852886e+38f)

typedef __bf16 bf16_t;
typedef bf16_t v16bf __attribute__((ext_vector_type(16)));
typedef float  v8f   __attribute__((ext_vector_type(8)));
typedef int    v4i   __attribute__((ext_vector_type(4)));
typedef int    v8i_t __attribute__((ext_vector_type(8)));

union FragBF { unsigned int u[8]; v16bf v; };

__device__ __forceinline__ unsigned short f2bf(float f) {
    unsigned int u = __float_as_uint(f);
    unsigned int lsb = (u >> 16) & 1u;
    u += 0x7fffu + lsb;               // round-to-nearest-even
    return (unsigned short)(u >> 16);
}

// CDNA5 async copy global->LDS (ASYNCcnt path, CDNA5 ISA)
__device__ __forceinline__ void async_ld_b128(unsigned lds_addr, const void* gaddr) {
    asm volatile("global_load_async_to_lds_b128 %0, %1, off"
                 :: "v"(lds_addr), "v"(gaddr) : "memory");
}
#define WAIT_ASYNC(n) asm volatile("s_wait_asynccnt " #n ::: "memory")

// CDNA5 Tensor Data Mover: 2-D tile DMA global -> LDS (D# groups 0+1)
__device__ __forceinline__ void tdm_load_2d(unsigned lds_addr, const void* gaddr,
                                            int tile_d0, int tile_d1,
                                            int tensor_d0, int tensor_d1,
                                            long long stride0_elems) {
    unsigned long long ga = (unsigned long long)gaddr;
    v4i g0; v8i_t g1;
    g0[0] = 1;                                            // count=1 (valid user D#)
    g0[1] = (int)lds_addr;                                // lds_addr (bytes)
    g0[2] = (int)(ga & 0xffffffffull);                    // global_addr[31:0]
    g0[3] = (int)((ga >> 32) & 0x1ffffffull) | (int)0x80000000u; // addr[56:32] | type=2
    g1[0] = 0x00010000;                                   // data_size=1 (2 bytes)
    g1[1] = (int)(((unsigned)tensor_d0 & 0xffffu) << 16); // tensor_dim0[15:0] @ bit48
    g1[2] = (int)((((unsigned)tensor_d0 >> 16) & 0xffffu) |
                  (((unsigned)tensor_d1 & 0xffffu) << 16));
    g1[3] = (int)((((unsigned)tensor_d1 >> 16) & 0xffffu) |
                  (((unsigned)tile_d0 & 0xffffu) << 16)); // tile_dim0 @ bit112
    g1[4] = (int)((unsigned)tile_d1 & 0xffffu);           // tile_dim1; tile_dim2=0
    g1[5] = (int)(stride0_elems & 0xffffffffll);          // tensor_dim0_stride lo
    g1[6] = (int)((stride0_elems >> 32) & 0xffffll);      // stride hi; dim1_stride=0
    g1[7] = 0;
    asm volatile("tensor_load_to_lds %0, %1" :: "s"(g0), "s"(g1) : "memory");
}

__device__ __forceinline__ float wave_sum(float v) {
#pragma unroll
    for (int o = 16; o > 0; o >>= 1) v += __shfl_xor(v, o, 32);
    return v;
}
__device__ __forceinline__ void wave_argmin(float& v, int& i) {
#pragma unroll
    for (int o = 16; o > 0; o >>= 1) {
        float ov = __shfl_xor(v, o, 32);
        int   oi = __shfl_xor(i, o, 32);
        if (ov < v || (ov == v && oi < i)) { v = ov; i = oi; }
    }
}

// ---------------- f32 -> bf16 bulk convert ----------------
__global__ void f32_to_bf16_k(const float* __restrict__ in,
                              unsigned short* __restrict__ out, int n) {
    int i = blockIdx.x * blockDim.x + threadIdx.x;
    if (i < n) out[i] = f2bf(in[i]);
}

// --------- C[M,N] = A[M,K] * B[N,K]^T  (bf16 in, f32 out) ---------------
// 8 waves / WG, 64x128 tile, 2x4 wave grid, 2x2 accums per wave.
// Double-buffered LDS staging via async global->LDS b128: next tile streams
// in (ASYNCcnt<=3 means current tile retired, in-order) while WMMAs run.
__global__ void __launch_bounds__(256)
gemm_nt_bf16(const unsigned short* __restrict__ A,
             const unsigned short* __restrict__ B,
             float* __restrict__ C,
             int K, int lda, int ldb, int ldc,
             long long strideA, long long strideB, long long strideC) {
    __shared__ unsigned short Asm[2][64][40];    // pad rows to 80B (16B aligned)
    __shared__ unsigned short Bsm[2][128][40];
    const int tm = blockIdx.x, tn = blockIdx.y, bb = blockIdx.z;
    const unsigned short* Ab = A + (long long)bb * strideA;
    const unsigned short* Bb = B + (long long)bb * strideB;
    float* Cb = C + (long long)bb * strideC;

    const int tid = threadIdx.x, wid = tid >> 5, lane = tid & 31;
    const int wm = wid >> 2, wn = wid & 3;      // 2 x 4 wave grid
    const int row16 = lane & 15, half = lane >> 4;

    // staging assignment: 8 bf16 (16B) per transfer
    const int srow = tid >> 2, sseg = (tid & 3) * 8;   // 64 rows x 4 segs
    const unsigned ldsA[2]  = { (unsigned)(size_t)&Asm[0][srow][sseg],
                                (unsigned)(size_t)&Asm[1][srow][sseg] };
    const unsigned ldsB0[2] = { (unsigned)(size_t)&Bsm[0][srow][sseg],
                                (unsigned)(size_t)&Bsm[1][srow][sseg] };
    const unsigned ldsB1[2] = { (unsigned)(size_t)&Bsm[0][srow + 64][sseg],
                                (unsigned)(size_t)&Bsm[1][srow + 64][sseg] };
    const unsigned short* gA  = Ab + (long long)(tm * 64 + srow) * lda + sseg;
    const unsigned short* gB0 = Bb + (long long)(tn * 128 + srow) * ldb + sseg;
    const unsigned short* gB1 = Bb + (long long)(tn * 128 + srow + 64) * ldb + sseg;

    const int nk = K >> 5;
    // prologue: stream tile 0
    async_ld_b128(ldsA[0],  gA);
    async_ld_b128(ldsB0[0], gB0);
    async_ld_b128(ldsB1[0], gB1);

    v8f acc[2][2] = {};
    for (int k = 0; k < nk; ++k) {
        __syncthreads();                       // prior reads of buf[(k+1)&1] done
        if (k + 1 < nk) {
            const int k0n = (k + 1) << 5, bi = (k + 1) & 1;
            async_ld_b128(ldsA[bi],  gA  + k0n);
            async_ld_b128(ldsB0[bi], gB0 + k0n);
            async_ld_b128(ldsB1[bi], gB1 + k0n);
            WAIT_ASYNC(0x3);                   // oldest 3 (tile k) retired
        } else {
            WAIT_ASYNC(0x0);
        }
        __syncthreads();                       // tile k visible to all waves

        const int bi = k & 1;
        FragBF a[2], b[2];
#pragma unroll
        for (int i = 0; i < 2; ++i) {
            int m = wm * 32 + i * 16 + row16;
#pragma unroll
            for (int v = 0; v < 8; ++v) {
                int ka = (v < 4 ? 2 * v : 2 * v + 8) + 8 * half;  // A 16x32 layout
                a[i].u[v] = *(const unsigned int*)&Asm[bi][m][ka];
            }
        }
#pragma unroll
        for (int j = 0; j < 2; ++j) {
            int n = wn * 32 + j * 16 + row16;
#pragma unroll
            for (int v = 0; v < 8; ++v) {
                int kb = (v < 4 ? 2 * v : 2 * v + 8) + 8 * half;  // B 32x16 layout
                b[j].u[v] = *(const unsigned int*)&Bsm[bi][n][kb];
            }
        }
#pragma unroll
        for (int i = 0; i < 2; ++i)
#pragma unroll
            for (int j = 0; j < 2; ++j)
                acc[i][j] = __builtin_amdgcn_wmma_f32_16x16x32_bf16(
                    false, a[i].v, false, b[j].v, (short)0, acc[i][j], false, false);
    }
#pragma unroll
    for (int i = 0; i < 2; ++i)
#pragma unroll
        for (int j = 0; j < 2; ++j)
#pragma unroll
            for (int r = 0; r < 8; ++r) {
                int m = tm * 64 + wm * 32 + i * 16 + r + 8 * half;
                int n = tn * 128 + wn * 32 + j * 16 + row16;
                Cb[(long long)m * ldc + n] = acc[i][j][r];
            }
}

// ---------------- RoPE + head-major repack (q,k) and V^T ----------------
__global__ void rope_pack_k(const float* __restrict__ qp, const float* __restrict__ kp,
                            const float* __restrict__ vp, const int* __restrict__ pos,
                            unsigned short* __restrict__ qh, unsigned short* __restrict__ kh,
                            unsigned short* __restrict__ vT) {
    int s = blockIdx.x, h = blockIdx.y, d = threadIdx.x;  // d in [0,128)
    int col = h * HDIM + d;
    float p = (float)pos[s];
    int i = d & 63;
    float inv_freq = __expf(-((float)(2 * i) / (float)HDIM) * __logf(10000.0f));
    float ang = p * inv_freq, sn, cs;
    __sincosf(ang, &sn, &cs);
    long long base = (long long)s * DMODEL;
    float q  = qp[base + col];
    float kk = kp[base + col];
    float qo = (d < 64) ? -qp[base + col + 64] : qp[base + col - 64];
    float ko = (d < 64) ? -kp[base + col + 64] : kp[base + col - 64];
    qh[((long long)h * S_LEN + s) * HDIM + d] = f2bf(q * cs + qo * sn);
    kh[((long long)h * S_LEN + s) * HDIM + d] = f2bf(kk * cs + ko * sn);
    vT[((long long)h * HDIM + d) * S_LEN + s] = f2bf(vp[base + col]);
}

// ------------- causal mask + clamp + row softmax (in place) -------------
__global__ void masked_softmax_k(float* __restrict__ scores) {
    long long rowid = blockIdx.x;               // h*S + i
    int i = (int)(rowid & (S_LEN - 1));
    float* p = scores + rowid * (long long)S_LEN;
    const float scale = 0.08838834764831845f;   // 1/sqrt(128)
    __shared__ float red[8];
    int tid = threadIdx.x, wv = tid >> 5, lane = tid & 31;
    float w[8];
    float lmax = -INFINITY;
#pragma unroll
    for (int e = 0; e < 8; ++e) {
        int j = tid + e * 256;
        float v = (j <= i) ? fmaxf(p[j] * scale, NEG_MIN_F) : NEG_MIN_F;
        w[e] = v;
        lmax = fmaxf(lmax, v);
    }
#pragma unroll
    for (int o = 16; o > 0; o >>= 1) lmax = fmaxf(lmax, __shfl_xor(lmax, o, 32));
    if (lane == 0) red[wv] = lmax;
    __syncthreads();
    float m = fmaxf(fmaxf(fmaxf(red[0], red[1]), fmaxf(red[2], red[3])),
                    fmaxf(fmaxf(red[4], red[5]), fmaxf(red[6], red[7])));
    __syncthreads();
    float ls = 0.f;
#pragma unroll
    for (int e = 0; e < 8; ++e) { float ev = __expf(w[e] - m); w[e] = ev; ls += ev; }
    ls = wave_sum(ls);
    if (lane == 0) red[wv] = ls;
    __syncthreads();
    float tot = red[0] + red[1] + red[2] + red[3] + red[4] + red[5] + red[6] + red[7];
    float inv = 1.0f / tot;
#pragma unroll
    for (int e = 0; e < 8; ++e) p[tid + e * 256] = w[e] * inv;
}

// ------------- sequential H2O greedy eviction scan (1 WG per head) -------
__global__ void __launch_bounds__(1024)
h2o_scan_k(const float* __restrict__ scores, unsigned int* __restrict__ maskw) {
    const int h = blockIdx.x, tid = threadIdx.x;
    const int wv = tid >> 5, lane = tid & 31;
    const int W = S_LEN / 32;
    __shared__ unsigned int mwsh[W];
    __shared__ float wred[32];
    __shared__ int   wredi[32];
    __shared__ float sh_inv;
    __shared__ int   sh_arg;
    const float* sh = scores + (long long)h * S_LEN * S_LEN;
    unsigned int* mw = maskw + (long long)h * S_LEN * W;

    const int s0 = tid, s1 = tid + 1024;
    if (tid < W) mwsh[tid] = 0xFFFFFFFFu;
    for (int e = tid; e < (CACHE + 1) * W; e += 1024) mw[e] = 0xFFFFFFFFu;

    float v0 = 0.f, v1 = 0.f;
    for (int i = 0; i < CACHE; ++i) {
        const float* r = sh + (long long)i * S_LEN;
        v0 = PENALTY_F * v0 + r[s0];
        v1 = PENALTY_F * v1 + r[s1];
    }
    __syncthreads();

    for (int t = CACHE; t <= S_LEN - 2; ++t) {
        const float* r = sh + (long long)t * S_LEN;
        float b0 = (float)((mwsh[s0 >> 5] >> (s0 & 31)) & 1u);
        float b1 = (float)((mwsh[s1 >> 5] >> (s1 & 31)) & 1u);
        float c0 = r[s0] * b0;
        float c1 = r[s1] * b1;
        float ls = wave_sum(c0 + c1);
        if (lane == 0) wred[wv] = ls;
        __syncthreads();
        if (wv == 0) {
            float x = wave_sum(wred[lane]);
            if (lane == 0) sh_inv = 1.0f / x;
        }
        __syncthreads();
        float inv = sh_inv;
        v0 = PENALTY_F * v0 + c0 * inv;
        v1 = PENALTY_F * v1 + c1 * inv;
        const int lim = t - RECENT;
        float bv = INFINITY; int bi = 0x7fffffff;
        if (s0 <= lim) { bv = v0; bi = s0; }
        if (s1 <= lim && (v1 < bv || (v1 == bv && s1 < bi))) { bv = v1; bi = s1; }
        wave_argmin(bv, bi);
        if (lane == 0) { wred[wv] = bv; wredi[wv] = bi; }
        __syncthreads();
        if (tid == 0) {
            float fv = wred[0]; int fi = wredi[0];
#pragma unroll
            for (int k = 1; k < 32; ++k) {
                float ov = wred[k]; int oi = wredi[k];
                if (ov < fv || (ov == fv && oi < fi)) { fv = ov; fi = oi; }
            }
            sh_arg = fi;
            mwsh[fi >> 5] &= ~(1u << (fi & 31));
        }
        __syncthreads();
        int fi = sh_arg;
        if (fi == s0) v0 = INFINITY;
        if (fi == s1) v1 = INFINITY;
        if (tid < W) mw[(long long)(t + 1) * W + tid] = mwsh[tid];
    }
}

// ------- probs = scores*mask/rowsum (bf16 tile) then probs @ V via WMMA --
// V tiles (32x128) DMA'd into LDS by the Tensor Data Mover (one wave issues,
// TENSORcnt gates completion) while all waves compute the probs tile.
__global__ void __launch_bounds__(256)
attn_v_k(const float* __restrict__ scores, const unsigned int* __restrict__ maskw,
         const unsigned short* __restrict__ vT, float* __restrict__ attnf) {
    const int qb = blockIdx.x;         // query block (16 rows)
    const int h  = blockIdx.y;
    const int tid = threadIdx.x;       // 256 = 8 waves
    const int wid = tid >> 5, lane = tid & 31;
    const int row16 = lane & 15, half = lane >> 4;
    const int W = S_LEN / 32;
    __shared__ float rs[16];
    __shared__ float rinv[16];
    __shared__ unsigned short At[16][34];   // probs tile, padded
    __shared__ unsigned short Vt[128][32];  // TDM destination: 128 hd rows x 32 keys

    const float* sc = scores + ((long long)h * S_LEN + (long long)qb * 16) * S_LEN;
    const unsigned int* mw = maskw + ((long long)h * S_LEN + (long long)qb * 16) * W;
    const unsigned short* gv = vT + (long long)h * HDIM * S_LEN;  // per-head [128][2048]
    const unsigned vt_lds = (unsigned)(size_t)&Vt[0][0];
    const int wave_uid = __builtin_amdgcn_readfirstlane(wid);     // scalar branch key

    if (tid < 16) rs[tid] = 0.f;
    __syncthreads();
    {   // row sums of scores*mask (16 threads per row), LDS float atomics
        int r = tid & 15, jt = tid >> 4;
        const float* pr = sc + (long long)r * S_LEN;
        const unsigned int* pm = mw + (long long)r * W;
        float part = 0.f;
        for (int j = jt; j < S_LEN; j += 16)
            part += pr[j] * (float)((pm[j >> 5] >> (j & 31)) & 1u);
        atomicAdd(&rs[r], part);
    }
    __syncthreads();
    if (tid < 16) rinv[tid] = 1.0f / rs[tid];
    __syncthreads();

    v8f acc = {};
    for (int k0 = 0; k0 < S_LEN; k0 += 32) {
        if (wave_uid == 0)   // single TDM per WG: DMA V[0:128][k0:k0+32] -> LDS
            tdm_load_2d(vt_lds, gv + k0, 32, HDIM, S_LEN, HDIM, S_LEN);
        // meanwhile: stage normalized probs tile 16x32 in bf16
        for (int e = tid; e < 512; e += 256) {
            int r = e >> 5, j = e & 31, ja = k0 + j;
            unsigned int bit = (mw[(long long)r * W + (ja >> 5)] >> (ja & 31)) & 1u;
            At[r][j] = f2bf(sc[(long long)r * S_LEN + ja] * (float)bit * rinv[r]);
        }
        if (wave_uid == 0) __builtin_amdgcn_s_wait_tensorcnt(0);
        __syncthreads();
        FragBF a, b;
#pragma unroll
        for (int v = 0; v < 8; ++v) {
            int ka = (v < 4 ? 2 * v : 2 * v + 8) + 8 * half;
            a.u[v] = *(const unsigned int*)&At[row16][ka];
            int kb = (v < 4 ? 2 * v : 2 * v + 8) + 8 * half;
            b.u[v] = *(const unsigned int*)&Vt[wid * 16 + row16][kb];
        }
        acc = __builtin_amdgcn_wmma_f32_16x16x32_bf16(false, a.v, false, b.v,
                                                      (short)0, acc, false, false);
        __syncthreads();
    }
#pragma unroll
    for (int r = 0; r < 8; ++r) {
        int m = qb * 16 + r + 8 * half;
        int col = h * HDIM + wid * 16 + row16;
        attnf[(long long)m * DMODEL + col] = acc[r];
    }
}

// ---------------------------------------------------------------------
extern "C" void kernel_launch(void* const* d_in, const int* in_sizes, int n_in,
                              void* d_out, int out_size, void* d_ws, size_t ws_size,
                              hipStream_t stream) {
    (void)in_sizes; (void)n_in; (void)out_size; (void)ws_size;
    const float* hs  = (const float*)d_in[0];
    const int*   pos = (const int*)d_in[2];
    const float* wq  = (const float*)d_in[3];
    const float* wk  = (const float*)d_in[4];
    const float* wv  = (const float*)d_in[5];
    const float* wo  = (const float*)d_in[6];
    float* out = (float*)d_out;

    char* base = (char*)d_ws;
    size_t off = 0;
    auto alloc = [&](size_t bytes) -> void* {
        void* p = base + off;
        off += (bytes + 255) & ~(size_t)255;
        return p;
    };
    const size_t MAT = (size_t)S_LEN * DMODEL;             // 4M elems
    unsigned short* hb  = (unsigned short*)alloc(MAT * 2);
    unsigned short* wqb = (unsigned short*)alloc(MAT * 2);
    unsigned short* wkb = (unsigned short*)alloc(MAT * 2);
    unsigned short* wvb = (unsigned short*)alloc(MAT * 2);
    unsigned short* wob = (unsigned short*)alloc(MAT * 2);
    float* qproj = (float*)alloc(MAT * 4);                 // reused later as attnf
    float* kproj = (float*)alloc(MAT * 4);
    float* vproj = (float*)alloc(MAT * 4);
    unsigned short* qh = (unsigned short*)alloc(MAT * 2);
    unsigned short* kh = (unsigned short*)alloc(MAT * 2);
    unsigned short* vT = (unsigned short*)alloc(MAT * 2);
    float* scoresb = (float*)alloc((size_t)NH * S_LEN * S_LEN * 4);   // 256 MB
    unsigned int* maskw = (unsigned int*)alloc((size_t)NH * S_LEN * (S_LEN / 32) * 4);
    unsigned short* attnb = (unsigned short*)alloc(MAT * 2);
    float* attnf = qproj;   // reuse: q projection dead after rope_pack

    const int n = (int)MAT;
    dim3 cvt((n + 255) / 256), cb(256);
    f32_to_bf16_k<<<cvt, cb, 0, stream>>>(hs, hb, n);
    f32_to_bf16_k<<<cvt, cb, 0, stream>>>(wq, wqb, n);
    f32_to_bf16_k<<<cvt, cb, 0, stream>>>(wk, wkb, n);
    f32_to_bf16_k<<<cvt, cb, 0, stream>>>(wv, wvb, n);
    f32_to_bf16_k<<<cvt, cb, 0, stream>>>(wo, wob, n);

    // QKV projections: [S,D] = hidden[S,D] @ W[D,D]^T
    dim3 gfull(S_LEN / 64, DMODEL / 128, 1);
    gemm_nt_bf16<<<gfull, 256, 0, stream>>>(hb, wqb, qproj, DMODEL, DMODEL, DMODEL, DMODEL, 0, 0, 0);
    gemm_nt_bf16<<<gfull, 256, 0, stream>>>(hb, wkb, kproj, DMODEL, DMODEL, DMODEL, DMODEL, 0, 0, 0);
    gemm_nt_bf16<<<gfull, 256, 0, stream>>>(hb, wvb, vproj, DMODEL, DMODEL, DMODEL, DMODEL, 0, 0, 0);

    rope_pack_k<<<dim3(S_LEN, NH), HDIM, 0, stream>>>(qproj, kproj, vproj, pos, qh, kh, vT);

    // per-head Q @ K^T (batched over heads via grid.z)
    gemm_nt_bf16<<<dim3(S_LEN / 64, S_LEN / 128, NH), 256, 0, stream>>>(
        qh, kh, scoresb, HDIM, HDIM, HDIM, S_LEN,
        (long long)S_LEN * HDIM, (long long)S_LEN * HDIM, (long long)S_LEN * S_LEN);

    masked_softmax_k<<<NH * S_LEN, 256, 0, stream>>>(scoresb);

    h2o_scan_k<<<NH, 1024, 0, stream>>>(scoresb, maskw);

    attn_v_k<<<dim3(S_LEN / 16, NH), 256, 0, stream>>>(scoresb, maskw, vT, attnf);

    f32_to_bf16_k<<<cvt, cb, 0, stream>>>(attnf, attnb, n);

    // output projection into d_out
    gemm_nt_bf16<<<gfull, 256, 0, stream>>>(attnb, wob, out, DMODEL, DMODEL, DMODEL, DMODEL, 0, 0, 0);
}
